// WorldModel_65506841198910
// MI455X (gfx1250) — matrix operand order repaired and checked by reading, not compiled, not run on hardware
//
#include <hip/hip_runtime.h>
#include <hip/hip_bf16.h>

typedef __bf16 bf16_t;
typedef __attribute__((ext_vector_type(16))) __bf16 v16bf;
typedef __attribute__((ext_vector_type(8)))  float  v8f;

#define BB   128
#define TT   64
#define OBSD 128
#define ACTD 32
#define LATD 256
#define DETD 1024
#define HIDD 1024

__device__ __forceinline__ float clipf(float x) {
    return fminf(fmaxf(x, -20.0f), 2.0f);
}

// ---------------------------------------------------------------------------
__global__ void zero_f32_kernel(float* __restrict__ p, int n) {
    int i = blockIdx.x * blockDim.x + threadIdx.x;
    if (i < n) p[i] = 0.0f;
}

// Convert row-major fp32 W[K][N] into WMMA-fragment-swizzled bf16.
// dst element i: e = i&15, lane = (i>>4)&31, tile = i>>9,
// kt = tile % kTiles, nTile = tile / kTiles,
// k = kt*32 + (lane>>4)*8 + (e&7) + (e>>3)*16, n = nTile*16 + (lane&15)   (zero-pad n >= N)
__global__ void cvt_swizzle_kernel(const float* __restrict__ W, bf16_t* __restrict__ dst,
                                   int N, int kTiles, int total) {
    int i = blockIdx.x * blockDim.x + threadIdx.x;
    if (i >= total) return;
    int e    = i & 15;
    int lane = (i >> 4) & 31;
    int tile = i >> 9;
    int kt   = tile % kTiles;
    int nTile = tile / kTiles;
    int k = kt * 32 + ((lane >> 4) << 3) + (e & 7) + ((e >> 3) << 4);
    int n = nTile * 16 + (lane & 15);
    float v = (n < N) ? W[(size_t)k * N + n] : 0.0f;
    dst[i] = (bf16_t)v;
}

// strided row-block copy (used for concatenation)
__global__ void copy_rows_kernel(const float* __restrict__ src, int sstride,
                                 float* __restrict__ dst, int dstride,
                                 int cols, int rows) {
    int i = blockIdx.x * blockDim.x + threadIdx.x;
    if (i >= rows * cols) return;
    int r = i / cols, c = i - r * cols;
    dst[(size_t)r * dstride + c] = src[(size_t)r * sstride + c];
}

// ---------------------------------------------------------------------------
// Pipeline helpers for the WMMA GEMM (KT = K/32 is compile-time)
template<int KT>
__device__ __forceinline__ void load_stage(const float* __restrict__ Arow, int g0, int kt,
                                           const bf16_t* __restrict__ Wj,
                                           float4* Af, v16bf* Bf) {
    const int k0 = kt * 32;
    Af[0] = *(const float4*)(Arow + k0 + g0);
    Af[1] = *(const float4*)(Arow + k0 + g0 + 4);
    Af[2] = *(const float4*)(Arow + k0 + 16 + g0);
    Af[3] = *(const float4*)(Arow + k0 + 16 + g0 + 4);
    const size_t koff = (size_t)kt * 512;
    Bf[0] = *(const v16bf*)(Wj + koff);
    Bf[1] = *(const v16bf*)(Wj + koff + (size_t)KT * 512);
    Bf[2] = *(const v16bf*)(Wj + koff + (size_t)KT * 1024);
    Bf[3] = *(const v16bf*)(Wj + koff + (size_t)KT * 1536);
}

__device__ __forceinline__ void compute_stage(const float4* Af, const v16bf* Bf, v8f* acc) {
    v16bf af;
    af[0]  = (bf16_t)Af[0].x; af[1]  = (bf16_t)Af[0].y; af[2]  = (bf16_t)Af[0].z; af[3]  = (bf16_t)Af[0].w;
    af[4]  = (bf16_t)Af[1].x; af[5]  = (bf16_t)Af[1].y; af[6]  = (bf16_t)Af[1].z; af[7]  = (bf16_t)Af[1].w;
    af[8]  = (bf16_t)Af[2].x; af[9]  = (bf16_t)Af[2].y; af[10] = (bf16_t)Af[2].z; af[11] = (bf16_t)Af[2].w;
    af[12] = (bf16_t)Af[3].x; af[13] = (bf16_t)Af[3].y; af[14] = (bf16_t)Af[3].z; af[15] = (bf16_t)Af[3].w;
    acc[0] = __builtin_amdgcn_wmma_f32_16x16x32_bf16(false, af, false, Bf[0], (short)0, acc[0], false, false);
    acc[1] = __builtin_amdgcn_wmma_f32_16x16x32_bf16(false, af, false, Bf[1], (short)0, acc[1], false, false);
    acc[2] = __builtin_amdgcn_wmma_f32_16x16x32_bf16(false, af, false, Bf[2], (short)0, acc[2], false, false);
    acc[3] = __builtin_amdgcn_wmma_f32_16x16x32_bf16(false, af, false, Bf[3], (short)0, acc[3], false, false);
}

// WMMA GEMM: C[128,Nout] = act( A[128,K](fp32,lda) * Wsw(bf16, fragment-swizzled) + bias )
// block = 128 threads = 4 waves; block tile 64(M) x 64(N); each wave: 16 x 64.
// Two-stage software pipeline with compile-time K; launch_bounds gives the
// allocator full VGPR headroom so the pipeline stages never spill.
// act: 0 = none, 1 = relu, 2 = elu
template<int KT>
__launch_bounds__(128, 1)
__global__ void gemm_bf16_kernel(const float* __restrict__ A, int lda,
                                 const bf16_t* __restrict__ Wsw,
                                 const float* __restrict__ bias,
                                 float* __restrict__ C, int ldc,
                                 int Nout, int act) {
    const int lane  = threadIdx.x & 31;
    const int wave  = threadIdx.x >> 5;
    const int nBase = blockIdx.x * 64;
    const int mBase = blockIdx.y * 64 + wave * 16;

    const int g0 = (lane >> 4) * 8;                       // A k-group offset
    const float* Arow = A + (size_t)(mBase + (lane & 15)) * lda;
    const bf16_t* Wj = Wsw + ((size_t)(nBase / 16) * KT) * 512 + lane * 16;

    v8f acc[4];
#pragma unroll
    for (int j = 0; j < 4; ++j) acc[j] = (v8f){0.f,0.f,0.f,0.f,0.f,0.f,0.f,0.f};

    float4 A0[4], A1[4];
    v16bf  B0[4], B1[4];

    load_stage<KT>(Arow, g0, 0, Wj, A0, B0);
    int kt = 0;
#pragma unroll
    for (; kt + 1 < KT; kt += 2) {
        load_stage<KT>(Arow, g0, kt + 1, Wj, A1, B1);
        compute_stage(A0, B0, acc);
        if (kt + 2 < KT)
            load_stage<KT>(Arow, g0, kt + 2, Wj, A0, B0);
        compute_stage(A1, B1, acc);
    }
    if (kt < KT)
        compute_stage(A0, B0, acc);

    const int rBase = mBase + ((lane >> 4) * 8);          // ISA C/D layout
#pragma unroll
    for (int j = 0; j < 4; ++j) {
        const int col = nBase + j * 16 + (lane & 15);
        if (col < Nout) {
            const float bb = bias[col];
#pragma unroll
            for (int r = 0; r < 8; ++r) {
                float v = acc[j][r] + bb;
                if (act == 1)      v = fmaxf(v, 0.0f);
                else if (act == 2) v = (v > 0.0f) ? v : (expf(v) - 1.0f);
                C[(size_t)(rBase + r) * ldc + col] = v;
            }
        }
    }
}

// ---------------------------------------------------------------------------
// GRU gate combine: h = (1-z)*n + z*h  (in place on h)
__global__ void gru_combine_kernel(const float* __restrict__ gi,
                                   const float* __restrict__ gh,
                                   float* __restrict__ h, int total, int D) {
    int i = blockIdx.x * blockDim.x + threadIdx.x;
    if (i >= total) return;
    int b = i / D, j = i - b * D;
    const float* gib = gi + (size_t)b * 3 * D;
    const float* ghb = gh + (size_t)b * 3 * D;
    float r = 1.0f / (1.0f + expf(-(gib[j]       + ghb[j])));
    float z = 1.0f / (1.0f + expf(-(gib[D + j]   + ghb[D + j])));
    float n = tanhf(gib[2 * D + j] + r * ghb[2 * D + j]);
    h[i] = (1.0f - z) * n + z * h[i];
}

// ---------------------------------------------------------------------------
// latent: clip, reparameterize, KL (stop_gradient => kl_bal fwd = relu(kl_raw-1))
__global__ void latent_kernel(const float* __restrict__ pstats,
                              const float* __restrict__ qstats,
                              const float* __restrict__ eps,
                              float* __restrict__ stoch,
                              float* __restrict__ accum, float invBT) {
    __shared__ float red[LATD];
    int b = blockIdx.x, j = threadIdx.x;
    float pm = pstats[(size_t)b * 2 * LATD + j];
    float pl = clipf(pstats[(size_t)b * 2 * LATD + LATD + j]);
    float qm = qstats[(size_t)b * 2 * LATD + j];
    float ql = clipf(qstats[(size_t)b * 2 * LATD + LATD + j]);
    stoch[(size_t)b * LATD + j] = qm + expf(ql) * eps[(size_t)b * LATD + j];
    float dm = qm - pm;
    red[j] = pl - ql + (expf(2.0f * ql) + dm * dm) * 0.5f * expf(-2.0f * pl) - 0.5f;
    __syncthreads();
    for (int s = LATD / 2; s > 0; s >>= 1) {
        if (j < s) red[j] += red[j + s];
        __syncthreads();
    }
    if (j == 0) {
        atomicAdd(&accum[3], red[0] * invBT);                      // kl_raw mean
        atomicAdd(&accum[2], fmaxf(red[0] - 1.0f, 0.0f) * invBT);  // kl_bal mean
    }
}

// ---------------------------------------------------------------------------
// decoder head outputs: om, olm, nll, mse
__global__ void decout_kernel(const float* __restrict__ dstats,
                              const float* __restrict__ obs_t, int obsStride,
                              float* __restrict__ om_out, int omStride,
                              float* __restrict__ olm_out, int olmStride,
                              float* __restrict__ accum, float invBT) {
    __shared__ float rmse[OBSD], rnll[OBSD], rol[OBSD];
    int b = blockIdx.x, j = threadIdx.x;
    float om = dstats[(size_t)b * 2 * OBSD + j];
    float ol = clipf(dstats[(size_t)b * 2 * OBSD + OBSD + j]);
    om_out[(size_t)b * omStride + j] = om;
    float err = om - obs_t[(size_t)b * obsStride + j];
    rmse[j] = err * err;
    rnll[j] = 0.5f * err * err * expf(-2.0f * ol) + ol + 0.91893853320467274f;
    rol[j]  = ol;
    __syncthreads();
    for (int s = OBSD / 2; s > 0; s >>= 1) {
        if (j < s) { rmse[j] += rmse[j + s]; rnll[j] += rnll[j + s]; rol[j] += rol[j + s]; }
        __syncthreads();
    }
    if (j == 0) {
        olm_out[(size_t)b * olmStride] = rol[0] * (1.0f / OBSD);
        atomicAdd(&accum[0], rnll[0] * (1.0f / OBSD) * invBT);  // nll mean
        atomicAdd(&accum[1], rmse[0] * (1.0f / OBSD) * invBT);  // mse mean
    }
}

// reward / continue heads
__global__ void rewcont_kernel(const float* __restrict__ rstats,
                               const float* __restrict__ cstats,
                               float* __restrict__ rm_out,
                               float* __restrict__ rl_out,
                               float* __restrict__ cl_out, int strideT) {
    int b = threadIdx.x;
    rm_out[(size_t)b * strideT] = rstats[b * 2];
    rl_out[(size_t)b * strideT] = clipf(rstats[b * 2 + 1]);
    cl_out[(size_t)b * strideT] = cstats[b];
}

__global__ void finalize_kernel(const float* __restrict__ accum, float* __restrict__ dst) {
    int i = threadIdx.x;
    if (i < 4) dst[i] = accum[i];
}

// ---------------------------------------------------------------------------
extern "C" void kernel_launch(void* const* d_in, const int* in_sizes, int n_in,
                              void* d_out, int out_size, void* d_ws, size_t ws_size,
                              hipStream_t stream) {
    (void)in_sizes; (void)n_in; (void)out_size; (void)ws_size;

    const float* obs     = (const float*)d_in[0];   // (B,T,128)
    const float* actions = (const float*)d_in[1];   // (B,T,32)
    const float* noise   = (const float*)d_in[2];   // (T,B,256)

    // params leaves (jax pytree: dict keys sorted, lists/tuples in order)
    const float* const* P = (const float* const*)(d_in + 3);
    const float *cont_W0=P[0], *cont_b0=P[1], *cont_W1=P[2], *cont_b1=P[3], *cont_W2=P[4], *cont_b2=P[5];
    const float *dec_W0 =P[6], *dec_b0 =P[7], *dec_W1 =P[8], *dec_b1 =P[9], *dec_W2 =P[10],*dec_b2 =P[11];
    const float *enc_W0 =P[12],*enc_b0 =P[13],*enc_W1 =P[14],*enc_b1 =P[15],*enc_W2 =P[16],*enc_b2 =P[17];
    const float *gru_Whh=P[18],*gru_Wih=P[19],*gru_bhh=P[20],*gru_bih=P[21];
    const float *inp_W  =P[22],*inp_b  =P[23];
    const float *post_W0=P[24],*post_b0=P[25],*post_W1=P[26],*post_b1=P[27];
    const float *pri_W0 =P[28],*pri_b0 =P[29],*pri_W1 =P[30],*pri_b1 =P[31];
    const float *rew_W0 =P[32],*rew_b0 =P[33],*rew_W1 =P[34],*rew_b1 =P[35],*rew_W2 =P[36],*rew_b2 =P[37];

    // ---- workspace carve-out ----
    char* wsc = (char*)d_ws;
    size_t cur = 0;
    auto alloc_bf = [&](size_t elems) -> bf16_t* {
        bf16_t* p = (bf16_t*)(wsc + cur); cur += ((elems * 2 + 255) / 256) * 256; return p; };
    auto alloc_f = [&](size_t elems) -> float* {
        float* p = (float*)(wsc + cur); cur += ((elems * 4 + 255) / 256) * 256; return p; };

    struct WCvt { const float* src; bf16_t* dst; int N; int kTiles; int total; };
    WCvt wc[19];
    int nw = 0;
    // K must be a multiple of 32; Npad multiple of 64 (zero-padded columns).
    auto mkw = [&](const float* s, int K, int N, int Npad) -> bf16_t* {
        int total = K * Npad;
        bf16_t* d = alloc_bf((size_t)total);
        wc[nw++] = {s, d, N, K / 32, total};
        return d; };

    const int XD = DETD + LATD + ACTD;   // 1312
    const int FD = DETD + LATD;          // 1280

    bf16_t* bEncW0 = mkw(enc_W0, OBSD, HIDD,     HIDD);
    bf16_t* bEncW1 = mkw(enc_W1, HIDD, HIDD,     HIDD);
    bf16_t* bEncW2 = mkw(enc_W2, HIDD, LATD,     LATD);
    bf16_t* bInpW  = mkw(inp_W,  XD,   HIDD,     HIDD);
    bf16_t* bWih   = mkw(gru_Wih,HIDD, 3 * DETD, 3 * DETD);
    bf16_t* bWhh   = mkw(gru_Whh,DETD, 3 * DETD, 3 * DETD);
    bf16_t* bPriW0 = mkw(pri_W0, DETD, HIDD,     HIDD);
    bf16_t* bPriW1 = mkw(pri_W1, HIDD, 2 * LATD, 2 * LATD);
    bf16_t* bPosW0 = mkw(post_W0,FD,   HIDD,     HIDD);
    bf16_t* bPosW1 = mkw(post_W1,HIDD, 2 * LATD, 2 * LATD);
    bf16_t* bDecW0 = mkw(dec_W0, FD,   HIDD,     HIDD);
    bf16_t* bDecW1 = mkw(dec_W1, HIDD, HIDD,     HIDD);
    bf16_t* bDecW2 = mkw(dec_W2, HIDD, 2 * OBSD, 2 * OBSD);
    bf16_t* bRewW0 = mkw(rew_W0, FD,   HIDD,     HIDD);
    bf16_t* bRewW1 = mkw(rew_W1, HIDD, HIDD,     HIDD);
    bf16_t* bRewW2 = mkw(rew_W2, HIDD, 2,        64);      // padded
    bf16_t* bConW0 = mkw(cont_W0,FD,   HIDD,     HIDD);
    bf16_t* bConW1 = mkw(cont_W1,HIDD, HIDD,     HIDD);
    bf16_t* bConW2 = mkw(cont_W2,HIDD, 1,        64);      // padded

    float* emb    = alloc_f((size_t)BB * LATD);
    float* hbuf1  = alloc_f((size_t)BB * HIDD);
    float* hbuf2  = alloc_f((size_t)BB * HIDD);
    float* xcat   = alloc_f((size_t)BB * XD);
    float* cell   = alloc_f((size_t)BB * HIDD);
    float* gi     = alloc_f((size_t)BB * 3 * DETD);
    float* gh     = alloc_f((size_t)BB * 3 * DETD);
    float* deter  = alloc_f((size_t)BB * DETD);
    float* stoch  = alloc_f((size_t)BB * LATD);
    float* pstats = alloc_f((size_t)BB * 2 * LATD);
    float* qin    = alloc_f((size_t)BB * FD);
    float* qstats = alloc_f((size_t)BB * 2 * LATD);
    float* dstats = alloc_f((size_t)BB * 2 * OBSD);
    float* rstats = alloc_f((size_t)BB * 2);
    float* cstats = alloc_f((size_t)BB * 1);
    float* zero32 = alloc_f(32);
    float* accum  = alloc_f(4);

    // ---- output offsets (return order) ----
    float* fout = (float*)d_out;
    const size_t FEAT_OFF = 0;
    const size_t OM_OFF   = FEAT_OFF + (size_t)BB * TT * FD;
    const size_t RM_OFF   = OM_OFF + (size_t)BB * TT * OBSD;
    const size_t RL_OFF   = RM_OFF + (size_t)BB * TT;
    const size_t OLM_OFF  = RL_OFF + (size_t)BB * TT;
    const size_t CL_OFF   = OLM_OFF + (size_t)BB * TT;
    const size_t SC_OFF   = CL_OFF + (size_t)BB * TT;

    const float invBT = 1.0f / (float)(BB * TT);

    auto zero = [&](float* p, int n) {
        zero_f32_kernel<<<(n + 255) / 256, 256, 0, stream>>>(p, n); };
    auto gemm = [&](const float* A, int lda, const bf16_t* W, const float* bias,
                    float* C, int ldc, int Nout, int Npad, int K, int act) {
        dim3 grid(Npad / 64, BB / 64);
        switch (K / 32) {
        case 4:
            gemm_bf16_kernel<4><<<grid, 128, 0, stream>>>(A, lda, W, bias, C, ldc, Nout, act);  break;
        case 32:
            gemm_bf16_kernel<32><<<grid, 128, 0, stream>>>(A, lda, W, bias, C, ldc, Nout, act); break;
        case 40:
            gemm_bf16_kernel<40><<<grid, 128, 0, stream>>>(A, lda, W, bias, C, ldc, Nout, act); break;
        case 41:
            gemm_bf16_kernel<41><<<grid, 128, 0, stream>>>(A, lda, W, bias, C, ldc, Nout, act); break;
        default: break;  // unused in this model
        }
    };
    auto crows = [&](const float* s, int ss, float* d, int ds, int cols, int rows) {
        copy_rows_kernel<<<(rows * cols + 255) / 256, 256, 0, stream>>>(s, ss, d, ds, cols, rows); };

    // ---- one-time (per call): weight conversion + state init ----
    for (int i = 0; i < nw; ++i)
        cvt_swizzle_kernel<<<(wc[i].total + 255) / 256, 256, 0, stream>>>(
            wc[i].src, wc[i].dst, wc[i].N, wc[i].kTiles, wc[i].total);
    zero(deter, BB * DETD);
    zero(stoch, BB * LATD);
    zero(zero32, 32);
    zero(accum, 4);

    for (int t = 0; t < TT; ++t) {
        const float* obs_t = obs + (size_t)t * OBSD;       // row stride T*OBSD

        // encoder MLP
        gemm(obs_t, TT * OBSD, bEncW0, enc_b0, hbuf1, HIDD, HIDD, HIDD, OBSD, 1);
        gemm(hbuf1, HIDD,      bEncW1, enc_b1, hbuf2, HIDD, HIDD, HIDD, HIDD, 1);
        gemm(hbuf2, HIDD,      bEncW2, enc_b2, emb,   LATD, LATD, LATD, HIDD, 0);

        // x = [deter | stoch | prev_act]
        crows(deter, DETD, xcat, XD, DETD, BB);
        crows(stoch, LATD, xcat + DETD, XD, LATD, BB);
        if (t == 0) crows(zero32, 0, xcat + DETD + LATD, XD, ACTD, BB);
        else        crows(actions + (size_t)(t - 1) * ACTD, TT * ACTD, xcat + DETD + LATD, XD, ACTD, BB);

        // cell input (ELU) + GRU
        gemm(xcat, XD,    bInpW, inp_b,   cell, HIDD,     HIDD,     HIDD,     XD,   2);
        gemm(cell, HIDD,  bWih,  gru_bih, gi,   3 * DETD, 3 * DETD, 3 * DETD, HIDD, 0);
        gemm(deter, DETD, bWhh,  gru_bhh, gh,   3 * DETD, 3 * DETD, 3 * DETD, DETD, 0);
        gru_combine_kernel<<<(BB * DETD + 255) / 256, 256, 0, stream>>>(gi, gh, deter, BB * DETD, DETD);

        // prior
        gemm(deter, DETD, bPriW0, pri_b0, hbuf1,  HIDD,     HIDD,     HIDD,     DETD, 1);
        gemm(hbuf1, HIDD, bPriW1, pri_b1, pstats, 2 * LATD, 2 * LATD, 2 * LATD, HIDD, 0);

        // posterior
        crows(deter, DETD, qin, FD, DETD, BB);
        crows(emb,   LATD, qin + DETD, FD, LATD, BB);
        gemm(qin,   FD,   bPosW0, post_b0, hbuf1,  HIDD,     HIDD,     HIDD,     FD,   1);
        gemm(hbuf1, HIDD, bPosW1, post_b1, qstats, 2 * LATD, 2 * LATD, 2 * LATD, HIDD, 0);

        // reparameterize + KL
        latent_kernel<<<BB, LATD, 0, stream>>>(pstats, qstats,
            noise + (size_t)t * BB * LATD, stoch, accum, invBT);

        // feat -> d_out (heads read it in place with lda = T*FD)
        float* featA = fout + FEAT_OFF + (size_t)t * FD;
        crows(deter, DETD, featA, TT * FD, DETD, BB);
        crows(stoch, LATD, featA + DETD, TT * FD, LATD, BB);

        // decoder head
        gemm(featA, TT * FD, bDecW0, dec_b0, hbuf1,  HIDD,     HIDD,     HIDD,     FD,   1);
        gemm(hbuf1, HIDD,    bDecW1, dec_b1, hbuf2,  HIDD,     HIDD,     HIDD,     HIDD, 1);
        gemm(hbuf2, HIDD,    bDecW2, dec_b2, dstats, 2 * OBSD, 2 * OBSD, 2 * OBSD, HIDD, 0);
        decout_kernel<<<BB, OBSD, 0, stream>>>(dstats, obs_t, TT * OBSD,
            fout + OM_OFF + (size_t)t * OBSD, TT * OBSD,
            fout + OLM_OFF + t, TT, accum, invBT);

        // reward head
        gemm(featA, TT * FD, bRewW0, rew_b0, hbuf1,  HIDD, HIDD, HIDD, FD,   1);
        gemm(hbuf1, HIDD,    bRewW1, rew_b1, hbuf2,  HIDD, HIDD, HIDD, HIDD, 1);
        gemm(hbuf2, HIDD,    bRewW2, rew_b2, rstats, 2,    2,    64,   HIDD, 0);

        // continue head
        gemm(featA, TT * FD, bConW0, cont_b0, hbuf1,  HIDD, HIDD, HIDD, FD,   1);
        gemm(hbuf1, HIDD,    bConW1, cont_b1, hbuf2,  HIDD, HIDD, HIDD, HIDD, 1);
        gemm(hbuf2, HIDD,    bConW2, cont_b2, cstats, 1,    1,    64,   HIDD, 0);

        rewcont_kernel<<<1, BB, 0, stream>>>(rstats, cstats,
            fout + RM_OFF + t, fout + RL_OFF + t, fout + CL_OFF + t, TT);
    }

    finalize_kernel<<<1, 4, 0, stream>>>(accum, fout + SC_OFF);
}
